// PrunedMultiHeadAttention_28046136442943
// MI455X (gfx1250) — matrix operand order repaired and checked by reading, not compiled, not run on hardware
//
#include <hip/hip_runtime.h>

// ---------------------------------------------------------------------------
// PrunedMultiHeadAttention for MI455X (gfx1250, wave32, WMMA bf16)
// B=16, N=1024, C=768, H=9, HD=64
// ---------------------------------------------------------------------------

typedef __attribute__((ext_vector_type(16))) __bf16 v16bf;
typedef __attribute__((ext_vector_type(8)))  float  v8f;

#define BDIM   16
#define NDIM   1024
#define CDIM   768
#define HDIM   9
#define HDHD   64
#define QKVN   (3 * HDIM * HDHD)   // 1728
#define CTXC   (HDIM * HDHD)       // 576
#define MROWS  (BDIM * NDIM)       // 16384
#define NHD    (NDIM * HDHD)       // 65536 elements per (which,b,h) slab

static __device__ __forceinline__ unsigned short f32_to_bf16(float f) {
  unsigned u = __float_as_uint(f);
  unsigned r = u + 0x7FFFu + ((u >> 16) & 1u);   // round-to-nearest-even
  return (unsigned short)(r >> 16);
}

// ---- WMMA fragment loaders (layouts per CDNA5 ISA 7.12.2) ------------------

// A-matrix 16x32 bf16 (row-major source, lda in elements, 16B-aligned rows).
// lane<16 holds row=lane, K = {half*8..+7} U {16+half*8..+7}
static __device__ __forceinline__ v16bf load_A(const __bf16* A, int lda, int lane) {
  int row = lane & 15, half = lane >> 4;
  const __bf16* p = A + (size_t)row * lda + half * 8;
  union { v16bf v; uint4 q[2]; } u;
  u.q[0] = *(const uint4*)(p);        // K = half*8 .. half*8+7
  u.q[1] = *(const uint4*)(p + 16);   // K = 16+half*8 .. 16+half*8+7
  return u.v;
}

// B-matrix 32x16 bf16 where the SOURCE is row-major [ncols][K] (i.e. we need
// its transpose as B).  B[kk][cc] = src[cc*ldk + kk] -> contiguous in kk.
// lane holds col=lane&15, K = {half*8..+7} U {16+half*8..+7} (matches load_A).
static __device__ __forceinline__ v16bf load_Bt(const __bf16* src, int ldk, int lane) {
  int col = lane & 15, half = lane >> 4;
  const __bf16* p = src + (size_t)col * ldk + half * 8;
  union { v16bf v; uint4 q[2]; } u;
  u.q[0] = *(const uint4*)(p);
  u.q[1] = *(const uint4*)(p + 16);
  return u.v;
}

static __device__ __forceinline__ v8f wmma_bf16(v16bf a, v16bf b, v8f c) {
  return __builtin_amdgcn_wmma_f32_16x16x32_bf16(
      /*neg_a=*/false, a, /*neg_b=*/false, b,
      /*c_mod=*/(short)0, c, /*reuse_a=*/false, /*reuse_b=*/false);
}

// ---------------------------------------------------------------------------
// Stage 0: fp32 -> bf16 conversion (vectorized 4-wide)
// ---------------------------------------------------------------------------
__global__ void cvt_f32_bf16(const float* __restrict__ s,
                             unsigned short* __restrict__ d, int n4) {
  int i = blockIdx.x * blockDim.x + threadIdx.x;
  if (i < n4) {
    float4 f = ((const float4*)s)[i];
    ushort4 o;
    o.x = f32_to_bf16(f.x); o.y = f32_to_bf16(f.y);
    o.z = f32_to_bf16(f.z); o.w = f32_to_bf16(f.w);
    ((ushort4*)d)[i] = o;
  }
}

// ---------------------------------------------------------------------------
// Stage 1: QKV GEMM  [16384 x 768] x [768 x 1728] + bias
// wave tile 32(M) x 64(N); 2-stage double-buffered K loop (no reg copies).
// Q scaled by HD^-0.5 at store; V stored transposed [b][h][d][n].
// ---------------------------------------------------------------------------
__global__ __launch_bounds__(128) void qkv_gemm_kernel(
    const __bf16* __restrict__ X,       // [16384][768]
    const __bf16* __restrict__ W,       // [1728][768] row-major (B = W^T)
    const float*  __restrict__ bias,    // [1728]
    unsigned short* __restrict__ qkv)   // [3][16][9][...] bf16
{
  const int lane = threadIdx.x & 31, wave = threadIdx.x >> 5;
  const int col = lane & 15, half = lane >> 4;
  const int m0 = blockIdx.x * 128 + wave * 32;
  const int n0 = blockIdx.y * 64;

  const __bf16* A0 = X + (size_t)m0 * CDIM;
  const __bf16* A1 = X + (size_t)(m0 + 16) * CDIM;
  const __bf16* Bp = W + (size_t)n0 * CDIM;   // 4 col-tiles at +t*16*CDIM

  v8f acc[2][4] = {};

  // even-stage fragments (k = 0)
  v16bf a0e = load_A(A0, CDIM, lane);
  v16bf a1e = load_A(A1, CDIM, lane);
  v16bf be[4];
#pragma unroll
  for (int t = 0; t < 4; ++t) be[t] = load_Bt(Bp + (size_t)t * 16 * CDIM, CDIM, lane);

  v16bf a0o, a1o, bo[4];

  for (int k = 0; k < CDIM; k += 64) {
    // odd-stage loads (k+32) issue while even-stage WMMAs run
    int k1 = k + 32;
    __builtin_prefetch(A0 + k + 512, 0, 1);   // global_prefetch_b8 (A stream)
    a0o = load_A(A0 + k1, CDIM, lane);
    a1o = load_A(A1 + k1, CDIM, lane);
#pragma unroll
    for (int t = 0; t < 4; ++t)
      bo[t] = load_Bt(Bp + (size_t)t * 16 * CDIM + k1, CDIM, lane);

#pragma unroll
    for (int t = 0; t < 4; ++t) {
      acc[0][t] = wmma_bf16(a0e, be[t], acc[0][t]);
      acc[1][t] = wmma_bf16(a1e, be[t], acc[1][t]);
    }

    // even-stage loads (k+64, wrap-safe) issue while odd-stage WMMAs run
    int k2 = (k + 64 < CDIM) ? k + 64 : 0;
    __builtin_prefetch(A1 + k + 512, 0, 1);
    a0e = load_A(A0 + k2, CDIM, lane);
    a1e = load_A(A1 + k2, CDIM, lane);
#pragma unroll
    for (int t = 0; t < 4; ++t)
      be[t] = load_Bt(Bp + (size_t)t * 16 * CDIM + k2, CDIM, lane);

#pragma unroll
    for (int t = 0; t < 4; ++t) {
      acc[0][t] = wmma_bf16(a0o, bo[t], acc[0][t]);
      acc[1][t] = wmma_bf16(a1o, bo[t], acc[1][t]);
    }
  }

#pragma unroll
  for (int t = 0; t < 4; ++t) {
    int gc = n0 + t * 16 + col;
    float bv = bias[gc];
    int which = gc / CTXC;
    int rem = gc - which * CTXC;
    int h = rem >> 6, dd = rem & 63;
#pragma unroll
    for (int mt = 0; mt < 2; ++mt) {
#pragma unroll
      for (int v = 0; v < 8; ++v) {
        int m = m0 + mt * 16 + v + 8 * half;
        int bb = m >> 10, nn = m & 1023;
        float val = acc[mt][t][v] + bv;
        if (which == 0) val *= 0.125f;  // 1/sqrt(64) folded into Q
        size_t base = ((size_t)((which * BDIM + bb) * HDIM + h)) * NHD;
        size_t idx = (which == 2) ? base + (size_t)dd * NDIM + nn   // V^T
                                  : base + (size_t)nn * HDHD + dd;  // Q,K
        qkv[idx] = f32_to_bf16(val);
      }
    }
  }
}

// ---------------------------------------------------------------------------
// Stage 2: flash attention.  One wave = one 16-row Q tile; online softmax;
// KV consumed in 32-key chunks, 2 chunks per iteration with alternating
// K-fragment buffers (loads overlap the other chunk's softmax VALU work).
// ---------------------------------------------------------------------------
__global__ __launch_bounds__(128) void attn_kernel(
    const __bf16* __restrict__ qkv,     // packed Q,K,V^T (bf16)
    unsigned short* __restrict__ ctx)   // [16384][576] bf16
{
  __shared__ alignas(16) unsigned short Pst[4][16 * 32];  // 1KB per wave

  const int lane = threadIdx.x & 31, wave = threadIdx.x >> 5;
  const int col = lane & 15, half = lane >> 4;
  const int bh = blockIdx.y;
  const int b = bh / HDIM, h = bh % HDIM;
  const int q0 = blockIdx.x * 64 + wave * 16;

  const __bf16* Q  = (const __bf16*)qkv + ((size_t)((0 * BDIM + b) * HDIM + h)) * NHD;
  const __bf16* K  = (const __bf16*)qkv + ((size_t)((1 * BDIM + b) * HDIM + h)) * NHD;
  const __bf16* Vt = (const __bf16*)qkv + ((size_t)((2 * BDIM + b) * HDIM + h)) * NHD;

  // Q tile 16x64 = two A fragments (already scaled by HD^-0.5)
  v16bf qa0 = load_A(Q + (size_t)q0 * HDHD +  0, HDHD, lane);
  v16bf qa1 = load_A(Q + (size_t)q0 * HDHD + 32, HDHD, lane);

  v8f accO[4] = {};
  float mrow[8], lrow[8];
#pragma unroll
  for (int v = 0; v < 8; ++v) { mrow[v] = -3.0e38f; lrow[v] = 0.0f; }

  unsigned short* P = &Pst[wave][0];
  const __bf16* Pb = (const __bf16*)P;

  auto loadK = [&](int kt, v16bf* kbf) {
    const __bf16* Kp0 = K + (size_t)kt * HDHD;
    const __bf16* Kp1 = K + (size_t)(kt + 16) * HDHD;
    kbf[0] = load_Bt(Kp0,      HDHD, lane);
    kbf[1] = load_Bt(Kp0 + 32, HDHD, lane);
    kbf[2] = load_Bt(Kp1,      HDHD, lane);
    kbf[3] = load_Bt(Kp1 + 32, HDHD, lane);
  };

  auto process = [&](int kt, const v16bf* kbf) {
    // ---- S chunk: 16 q-rows x 32 keys (two 16x16 C tiles) ----
    v8f s0 = {}, s1 = {};
    s0 = wmma_bf16(qa0, kbf[0], s0);
    s0 = wmma_bf16(qa1, kbf[1], s0);
    s1 = wmma_bf16(qa0, kbf[2], s1);
    s1 = wmma_bf16(qa1, kbf[3], s1);

    // ---- online softmax update, write P (bf16) to LDS in C layout ----
#pragma unroll
    for (int v = 0; v < 8; ++v) {
      float a = s0[v], c = s1[v];
      float mx = fmaxf(a, c);
#pragma unroll
      for (int off = 1; off < 16; off <<= 1)
        mx = fmaxf(mx, __shfl_xor(mx, off, 32));   // stays within 16-lane half
      float newm = fmaxf(mrow[v], mx);
      float p0 = __expf(a - newm);
      float p1 = __expf(c - newm);
      float rs = p0 + p1;
#pragma unroll
      for (int off = 1; off < 16; off <<= 1)
        rs += __shfl_xor(rs, off, 32);
      float alpha = __expf(mrow[v] - newm);        // exp(-inf)=0 on first chunk
      lrow[v] = lrow[v] * alpha + rs;
      mrow[v] = newm;
      accO[0][v] *= alpha; accO[1][v] *= alpha;
      accO[2][v] *= alpha; accO[3][v] *= alpha;
      int r = v + 8 * half;
      P[r * 32 + col]      = f32_to_bf16(p0);
      P[r * 32 + 16 + col] = f32_to_bf16(p1);
    }

    // ---- re-read P as A fragment (C->A relayout through LDS) ----
    v16bf pa = load_A(Pb, 32, lane);

    // ---- O += P * V  (V^T layout makes B loads contiguous) ----
#pragma unroll
    for (int t = 0; t < 4; ++t) {
      v16bf vb = load_Bt(Vt + (size_t)(t * 16) * NDIM + kt, NDIM, lane);
      accO[t] = wmma_bf16(pa, vb, accO[t]);
    }
  };

  v16bf kbe[4], kbo[4];
  loadK(0, kbe);
  for (int kt = 0; kt < NDIM; kt += 64) {
    loadK(kt + 32, kbo);                           // overlaps chunk kt
    process(kt, kbe);
    loadK((kt + 64 < NDIM) ? kt + 64 : 0, kbe);    // overlaps chunk kt+32
    process(kt + 32, kbo);
  }

  // ---- epilogue: normalize and store context [b][n][h*64+d] ----
  float invl[8];
#pragma unroll
  for (int v = 0; v < 8; ++v) invl[v] = 1.0f / lrow[v];
#pragma unroll
  for (int t = 0; t < 4; ++t) {
#pragma unroll
    for (int v = 0; v < 8; ++v) {
      int row = q0 + v + 8 * half;
      int d = t * 16 + col;
      float val = accO[t][v] * invl[v];
      ctx[((size_t)(b * NDIM + row)) * CTXC + h * HDHD + d] = f32_to_bf16(val);
    }
  }
}

// ---------------------------------------------------------------------------
// Stage 3: output projection  [16384 x 576] x [576 x 768] + bias -> fp32
// wave tile 32(M) x 64(N); 2-stage double-buffered K loop.
// ---------------------------------------------------------------------------
__global__ __launch_bounds__(128) void proj_gemm_kernel(
    const __bf16* __restrict__ CTX,     // [16384][576]
    const __bf16* __restrict__ PW,      // [768][576] row-major (B = PW^T)
    const float*  __restrict__ bias,    // [768]
    float* __restrict__ out)            // [16384][768]
{
  const int lane = threadIdx.x & 31, wave = threadIdx.x >> 5;
  const int col = lane & 15, half = lane >> 4;
  const int m0 = blockIdx.x * 128 + wave * 32;
  const int n0 = blockIdx.y * 64;

  const __bf16* A0 = CTX + (size_t)m0 * CTXC;
  const __bf16* A1 = CTX + (size_t)(m0 + 16) * CTXC;
  const __bf16* Bp = PW + (size_t)n0 * CTXC;

  v8f acc[2][4] = {};

  v16bf a0e = load_A(A0, CTXC, lane);
  v16bf a1e = load_A(A1, CTXC, lane);
  v16bf be[4];
#pragma unroll
  for (int t = 0; t < 4; ++t) be[t] = load_Bt(Bp + (size_t)t * 16 * CTXC, CTXC, lane);

  v16bf a0o, a1o, bo[4];

  for (int k = 0; k < CTXC; k += 64) {
    int k1 = k + 32;
    __builtin_prefetch(A0 + k + 512, 0, 1);
    a0o = load_A(A0 + k1, CTXC, lane);
    a1o = load_A(A1 + k1, CTXC, lane);
#pragma unroll
    for (int t = 0; t < 4; ++t)
      bo[t] = load_Bt(Bp + (size_t)t * 16 * CTXC + k1, CTXC, lane);

#pragma unroll
    for (int t = 0; t < 4; ++t) {
      acc[0][t] = wmma_bf16(a0e, be[t], acc[0][t]);
      acc[1][t] = wmma_bf16(a1e, be[t], acc[1][t]);
    }

    int k2 = (k + 64 < CTXC) ? k + 64 : 0;
    __builtin_prefetch(A1 + k + 512, 0, 1);
    a0e = load_A(A0 + k2, CTXC, lane);
    a1e = load_A(A1 + k2, CTXC, lane);
#pragma unroll
    for (int t = 0; t < 4; ++t)
      be[t] = load_Bt(Bp + (size_t)t * 16 * CTXC + k2, CTXC, lane);

#pragma unroll
    for (int t = 0; t < 4; ++t) {
      acc[0][t] = wmma_bf16(a0o, bo[t], acc[0][t]);
      acc[1][t] = wmma_bf16(a1o, bo[t], acc[1][t]);
    }
  }

#pragma unroll
  for (int t = 0; t < 4; ++t) {
    int gc = n0 + t * 16 + col;
    float bv = bias[gc];
#pragma unroll
    for (int mt = 0; mt < 2; ++mt) {
#pragma unroll
      for (int v = 0; v < 8; ++v) {
        int m = m0 + mt * 16 + v + 8 * half;
        out[(size_t)m * CDIM + gc] = acc[mt][t][v] + bv;
      }
    }
  }
}

// ---------------------------------------------------------------------------
// Host launcher
// ---------------------------------------------------------------------------
extern "C" void kernel_launch(void* const* d_in, const int* in_sizes, int n_in,
                              void* d_out, int out_size, void* d_ws, size_t ws_size,
                              hipStream_t stream) {
  const float* x      = (const float*)d_in[0];
  const float* qkv_w  = (const float*)d_in[1];
  const float* qkv_b  = (const float*)d_in[2];
  const float* proj_w = (const float*)d_in[3];
  const float* proj_b = (const float*)d_in[4];
  float* out = (float*)d_out;

  // workspace layout (bytes), all 256-aligned; total ~99.4 MB
  char* ws = (char*)d_ws;
  const size_t SZ_XB  = (size_t)MROWS * CDIM * 2;          // 25,165,824
  const size_t SZ_WQ  = (size_t)QKVN * CDIM * 2;           //  2,654,208
  const size_t SZ_PW  = (size_t)CDIM * CTXC * 2;           //    884,736
  const size_t SZ_QKV = (size_t)3 * BDIM * HDIM * NHD * 2; // 56,623,104
  unsigned short* Xb   = (unsigned short*)(ws);
  unsigned short* Wq   = (unsigned short*)(ws + SZ_XB);
  unsigned short* Pw   = (unsigned short*)(ws + SZ_XB + SZ_WQ);
  unsigned short* QKV  = (unsigned short*)(ws + SZ_XB + SZ_WQ + SZ_PW);
  unsigned short* CTXb = (unsigned short*)(ws + SZ_XB + SZ_WQ + SZ_PW + SZ_QKV);

  // Stage 0: fp32 -> bf16
  {
    int n4;
    n4 = (MROWS * CDIM) / 4;
    cvt_f32_bf16<<<(n4 + 255) / 256, 256, 0, stream>>>(x, Xb, n4);
    n4 = (QKVN * CDIM) / 4;
    cvt_f32_bf16<<<(n4 + 255) / 256, 256, 0, stream>>>(qkv_w, Wq, n4);
    n4 = (CDIM * CTXC) / 4;
    cvt_f32_bf16<<<(n4 + 255) / 256, 256, 0, stream>>>(proj_w, Pw, n4);
  }

  // Stage 1: QKV projection (block tile 128x64)
  qkv_gemm_kernel<<<dim3(MROWS / 128, QKVN / 64), 128, 0, stream>>>(
      (const __bf16*)Xb, (const __bf16*)Wq, qkv_b, QKV);

  // Stage 2: attention
  attn_kernel<<<dim3(NDIM / 64, BDIM * HDIM), 128, 0, stream>>>(
      (const __bf16*)QKV, CTXb);

  // Stage 3: output projection (block tile 128x64)
  proj_gemm_kernel<<<dim3(MROWS / 128, CDIM / 64), 128, 0, stream>>>(
      (const __bf16*)CTXb, (const __bf16*)Pw, proj_b, out);
}
